// ROIHeads_71854802862375
// MI455X (gfx1250) — compile-verified, matrix operand order checked by reading, not run on hardware
//
#include <hip/hip_runtime.h>
#include <hip/hip_bf16.h>

// ---------------------------------------------------------------- types
typedef __attribute__((ext_vector_type(8)))  __bf16 v8bf;
typedef __attribute__((ext_vector_type(16))) __bf16 v16bf;
typedef __attribute__((ext_vector_type(8)))  float  v8f;
typedef __attribute__((ext_vector_type(4)))  int    ai4;   // raw int4 vector for async builtin

__device__ __forceinline__ unsigned short f2bf(float f) {
    unsigned u = __float_as_uint(f);
    u += 0x7fffu + ((u >> 16) & 1u);      // round-to-nearest-even
    return (unsigned short)(u >> 16);
}

__device__ __forceinline__ v16bf cat16(v8bf lo, v8bf hi) {
    return __builtin_shufflevector(lo, hi, 0,1,2,3,4,5,6,7,8,9,10,11,12,13,14,15);
}

// CDNA5 async global->LDS staging (guarded; falls back to sync staging)
#if defined(__gfx1250__) && __has_builtin(__builtin_amdgcn_global_load_async_to_lds_b128)
#define HAVE_ASYNC_LDS 1
#else
#define HAVE_ASYNC_LDS 0
#endif

// ---------------------------------------------------------------- constants
#define NIMG   2
#define MPROP  1000
#define NROWS  (NIMG*MPROP)          // 2000
#define CCH    256
#define RPOOL  7
#define FCIN   (CCH*RPOOL*RPOOL)     // 12544
#define FCDIM  1024
#define SCALE_CLAMP 4.135166556742356f
#define SCORE_THRESH 0.05f
#define NMS_THRESH   0.5f
#define TOPK 100

// ---------------------------------------------------------------- f32 (K,N) -> bf16 bits transposed (N,K)
__global__ void cvt_transpose_bf16_kernel(const float* __restrict__ src,
                                          unsigned short* __restrict__ dst,
                                          int K, int N) {
    int i = blockIdx.x * blockDim.x + threadIdx.x;
    if (i < K * N) {
        int k = i / N, n = i - k * N;
        dst[(size_t)n * K + k] = f2bf(src[i]);
    }
}

// ---------------------------------------------------------------- ROI pooling (gather) -> bf16 feature rows
__global__ void pool_kernel(const float* __restrict__ p2, const float* __restrict__ p3,
                            const float* __restrict__ p4, const float* __restrict__ p5,
                            const float* __restrict__ proposals,
                            unsigned short* __restrict__ Xbf) {
    const int roi = blockIdx.x;                 // 0..1999
    const int img = roi / MPROP;
    const float* bx = proposals + (size_t)roi * 4;
    float x1 = bx[0], y1 = bx[1], x2 = bx[2], y2 = bx[3];

    float w = x2 - x1, h = y2 - y1;
    float s = sqrtf(fmaxf(w * h, 1e-6f));
    float lv = floorf(4.0f + log2f(s / 224.0f + 1e-8f));
    lv = fminf(fmaxf(lv, 2.0f), 5.0f);
    int lvl = (int)lv - 2;                      // 0..3

    const float scl[4] = {0.25f, 0.125f, 0.0625f, 0.03125f};
    const int   Hs[4]  = {200, 100, 50, 25};
    const int   Ws[4]  = {304, 152, 76, 38};
    const float* fmap = (lvl == 0 ? p2 : lvl == 1 ? p3 : lvl == 2 ? p4 : p5);
    const int Hc = Hs[lvl], Wd = Ws[lvl];
    fmap += (size_t)img * CCH * Hc * Wd;

    float sc = scl[lvl];
    float bx1 = x1 * sc, by1 = y1 * sc, bx2 = x2 * sc, by2 = y2 * sc;
    float bw = fmaxf(bx2 - bx1, 1.0f) / (float)RPOOL;
    float bh = fmaxf(by2 - by1, 1.0f) / (float)RPOOL;

    int ix[RPOOL], iy[RPOOL];
#pragma unroll
    for (int c = 0; c < RPOOL; ++c) {
        int vx = (int)(bx1 + ((float)c + 0.5f) * bw);
        int vy = (int)(by1 + ((float)c + 0.5f) * bh);
        ix[c] = min(max(vx, 0), Wd - 1);
        iy[c] = min(max(vy, 0), Hc - 1);
    }

    const int ch = threadIdx.x;                 // one channel per thread
    const float* fc = fmap + (size_t)ch * Hc * Wd;
    unsigned short* dst = Xbf + (size_t)roi * FCIN + ch * (RPOOL * RPOOL);
#pragma unroll
    for (int py = 0; py < RPOOL; ++py)
#pragma unroll
        for (int px = 0; px < RPOOL; ++px)
            dst[py * RPOOL + px] = f2bf(fc[iy[py] * Wd + ix[px]]);
}

// ---------------------------------------------------------------- per-wave 32x64 WMMA compute (8 x v_wmma 16x16x32 bf16)
__device__ __forceinline__ void wave_compute(const unsigned short* Asl, const unsigned short* Bsl,
                                             int wm, int wn, int half, int l16, v8f acc[2][4]) {
    v16bf aF[2], bF[4];
#pragma unroll
    for (int ti = 0; ti < 2; ++ti) {
        const v8bf* Ap = (const v8bf*)(Asl + (wm * 32 + ti * 16 + l16) * 32);
        aF[ti] = cat16(Ap[half], Ap[2 + half]);          // K runs at half*8 and 16+half*8
    }
#pragma unroll
    for (int tj = 0; tj < 4; ++tj) {
        const v8bf* Bp = (const v8bf*)(Bsl + (wn * 64 + tj * 16 + l16) * 32);
        bF[tj] = cat16(Bp[half], Bp[2 + half]);
    }
#pragma unroll
    for (int ti = 0; ti < 2; ++ti)
#pragma unroll
        for (int tj = 0; tj < 4; ++tj)
            acc[ti][tj] = __builtin_amdgcn_wmma_f32_16x16x32_bf16(
                false, aF[ti], false, bF[tj], (short)0, acc[ti][tj], false, false);
}

// ---------------------------------------------------------------- bf16 WMMA GEMM + bias + relu
//   C[M,N] = relu(A[M,K] * Bt[N,K]^T + bias[N]); A row-major, Bt = pre-transposed weights (N x K).
//   Block tile 128x128, 8 waves (4 M x 2 N), each wave 32x64 via 2x4 wmma.
//   Async double-buffered LDS pipeline when GLOBAL_LOAD_ASYNC_TO_LDS is available.
__global__ __launch_bounds__(256)
void gemm_bf16_wmma(const unsigned short* __restrict__ A,
                    const unsigned short* __restrict__ Btg,
                    const float* __restrict__ bias,
                    float* __restrict__ Cf,                 // optional f32 out
                    unsigned short* __restrict__ Cbf,       // optional bf16 out
                    int M, int N, int K) {
    __shared__ __align__(16) unsigned short Asm[2][128 * 32];
    __shared__ __align__(16) unsigned short Bsm[2][128 * 32];

    const int t    = threadIdx.x;
    const int bm   = blockIdx.x * 128;
    const int bn   = blockIdx.y * 128;
    const int wid  = t >> 5;
    const int lane = t & 31;
    const int wm   = wid & 3;        // 4 wave-rows (32 rows each)
    const int wn   = wid >> 2;       // 2 wave-cols (64 cols each)
    const int half = lane >> 4;
    const int l16  = lane & 15;

    v8f acc[2][4] = {};

    // staging geometry: each thread owns 32B of a 64B tile row (128 rows x 32 ushort)
    const int row2 = t >> 1, seg = t & 1;
    const int arow = min(bm + row2, M - 1);    // clamp OOB rows (results discarded in epilogue)
    const int brow = bn + row2;
    const unsigned short* gA = A   + (size_t)arow * K + seg * 16;
    const unsigned short* gB = Btg + (size_t)brow * K + seg * 16;
    const int lofs = row2 * 32 + seg * 16;
    const int nk = K >> 5;

#if HAVE_ASYNC_LDS
    // builtin wants int4-vector pointers: global (AS1) source, LDS (AS3) dest
    #define GAS(p) ((__attribute__((address_space(1))) ai4*)(p))
    #define LAS(p) ((__attribute__((address_space(3))) ai4*)(p))
    // prologue: async-stage k-slice 0 into buffer 0 (two b128 per thread per tile)
    __builtin_amdgcn_global_load_async_to_lds_b128(GAS(gA), LAS(&Asm[0][lofs]), 0, 0);
    __builtin_amdgcn_global_load_async_to_lds_b128(GAS(gA), LAS(&Asm[0][lofs]), 16, 0);
    __builtin_amdgcn_global_load_async_to_lds_b128(GAS(gB), LAS(&Bsm[0][lofs]), 0, 0);
    __builtin_amdgcn_global_load_async_to_lds_b128(GAS(gB), LAS(&Bsm[0][lofs]), 16, 0);

    for (int i = 0; i < nk; ++i) {
#if __has_builtin(__builtin_amdgcn_s_wait_asynccnt)
        __builtin_amdgcn_s_wait_asynccnt(0);
#else
        asm volatile("s_wait_asynccnt 0x0" ::: "memory");
#endif
        __syncthreads();                       // this wave's copies done + all waves' copies visible
        if (i + 1 < nk) {                      // issue next slice into the other buffer
            const unsigned short* gA2 = gA + (size_t)(i + 1) * 32;
            const unsigned short* gB2 = gB + (size_t)(i + 1) * 32;
            int nb = (i + 1) & 1;
            __builtin_amdgcn_global_load_async_to_lds_b128(GAS(gA2), LAS(&Asm[nb][lofs]), 0, 0);
            __builtin_amdgcn_global_load_async_to_lds_b128(GAS(gA2), LAS(&Asm[nb][lofs]), 16, 0);
            __builtin_amdgcn_global_load_async_to_lds_b128(GAS(gB2), LAS(&Bsm[nb][lofs]), 0, 0);
            __builtin_amdgcn_global_load_async_to_lds_b128(GAS(gB2), LAS(&Bsm[nb][lofs]), 16, 0);
        }
        wave_compute(&Asm[i & 1][0], &Bsm[i & 1][0], wm, wn, half, l16, acc);
    }
    #undef GAS
    #undef LAS
#else
    // synchronous fallback: global -> VGPR -> LDS, single buffer
    for (int i = 0; i < nk; ++i) {
        const uint4* ga = (const uint4*)(gA + (size_t)i * 32);
        uint4 a0 = ga[0], a1 = ga[1];
        const uint4* gb = (const uint4*)(gB + (size_t)i * 32);
        uint4 b0 = gb[0], b1 = gb[1];
        if (i + 2 < nk) {
            __builtin_prefetch(ga + 4, 0, 1);
            __builtin_prefetch(gb + 4, 0, 1);
        }
        uint4* la = (uint4*)(&Asm[0][lofs]); la[0] = a0; la[1] = a1;
        uint4* lb = (uint4*)(&Bsm[0][lofs]); lb[0] = b0; lb[1] = b1;
        __syncthreads();
        wave_compute(&Asm[0][0], &Bsm[0][0], wm, wn, half, l16, acc);
        __syncthreads();
    }
#endif

    // ---- epilogue: bias + relu, f32 and/or bf16 store
#pragma unroll
    for (int ti = 0; ti < 2; ++ti)
#pragma unroll
        for (int tj = 0; tj < 4; ++tj) {
            int gn = bn + wn * 64 + tj * 16 + l16;
            float bv = bias ? bias[gn] : 0.0f;
#pragma unroll
            for (int r = 0; r < 8; ++r) {
                int gm = bm + wm * 32 + ti * 16 + r + half * 8;
                if (gm < M) {
                    float v = fmaxf(acc[ti][tj][r] + bv, 0.0f);
                    if (Cf)  Cf [(size_t)gm * N + gn] = v;
                    if (Cbf) Cbf[(size_t)gm * N + gn] = f2bf(v);
                }
            }
        }
}

// ---------------------------------------------------------------- heads: cls/box GEMV + softmax + delta decode + clip
__global__ void heads_kernel(const float* __restrict__ X2,
                             const float* __restrict__ Wc, const float* __restrict__ bc,
                             const float* __restrict__ Wb, const float* __restrict__ bb,
                             const float* __restrict__ proposals,
                             const int*   __restrict__ image_sizes,
                             float* __restrict__ pboxes, float* __restrict__ pscores) {
    const int wid  = threadIdx.x >> 5;
    const int lane = threadIdx.x & 31;
    const int row  = blockIdx.x * 8 + wid;
    if (row >= NROWS) return;

    const float* x = X2 + (size_t)row * FCDIM;
    float c0 = 0.f, c1 = 0.f, d0 = 0.f, d1 = 0.f, d2 = 0.f, d3 = 0.f;
    for (int k = lane; k < FCDIM; k += 32) {
        float xv = x[k];
        c0 += xv * Wc[k * 2 + 0];
        c1 += xv * Wc[k * 2 + 1];
        d0 += xv * Wb[k * 4 + 0];
        d1 += xv * Wb[k * 4 + 1];
        d2 += xv * Wb[k * 4 + 2];
        d3 += xv * Wb[k * 4 + 3];
    }
#pragma unroll
    for (int off = 16; off > 0; off >>= 1) {
        c0 += __shfl_xor(c0, off, 32);
        c1 += __shfl_xor(c1, off, 32);
        d0 += __shfl_xor(d0, off, 32);
        d1 += __shfl_xor(d1, off, 32);
        d2 += __shfl_xor(d2, off, 32);
        d3 += __shfl_xor(d3, off, 32);
    }
    if (lane == 0) {
        c0 += bc[0]; c1 += bc[1];
        d0 += bb[0]; d1 += bb[1]; d2 += bb[2]; d3 += bb[3];
        float score = 1.0f / (1.0f + __expf(c0 - c1));    // softmax class-1 prob

        const float* p = proposals + (size_t)row * 4;
        float w = p[2] - p[0], h = p[3] - p[1];
        float cx = p[0] + 0.5f * w, cy = p[1] + 0.5f * h;
        float dw = fminf(d2, SCALE_CLAMP), dh = fminf(d3, SCALE_CLAMP);
        float pcx = d0 * w + cx, pcy = d1 * h + cy;
        float pw = __expf(dw) * w, ph = __expf(dh) * h;

        int img = row / MPROP;
        float Hf = (float)image_sizes[img * 2 + 0];
        float Wf = (float)image_sizes[img * 2 + 1];
        float bx1 = fminf(fmaxf(pcx - 0.5f * pw, 0.f), Wf);
        float by1 = fminf(fmaxf(pcy - 0.5f * ph, 0.f), Hf);
        float bx2 = fminf(fmaxf(pcx + 0.5f * pw, 0.f), Wf);
        float by2 = fminf(fmaxf(pcy + 0.5f * ph, 0.f), Hf);

        pboxes[row * 4 + 0] = bx1;
        pboxes[row * 4 + 1] = by1;
        pboxes[row * 4 + 2] = bx2;
        pboxes[row * 4 + 3] = by2;
        pscores[row] = score;
    }
}

// ---------------------------------------------------------------- rank sort (descending score, stable)
__global__ void sort_rank_kernel(const float* __restrict__ pboxes,
                                 const float* __restrict__ pscores,
                                 float* __restrict__ sboxes, float* __restrict__ sscores,
                                 float* __restrict__ outOrder) {
    int i = blockIdx.x * blockDim.x + threadIdx.x;
    if (i >= NROWS) return;
    int img  = i / MPROP;
    int li   = i - img * MPROP;
    int base = img * MPROP;
    float si = pscores[i];
    int rank = 0;
    for (int j = 0; j < MPROP; ++j) {
        float sj = pscores[base + j];
        rank += (sj > si) || (sj == si && j < li);
    }
    int o = base + rank;
    sscores[o] = si;
    sboxes[o * 4 + 0] = pboxes[i * 4 + 0];
    sboxes[o * 4 + 1] = pboxes[i * 4 + 1];
    sboxes[o * 4 + 2] = pboxes[i * 4 + 2];
    sboxes[o * 4 + 3] = pboxes[i * 4 + 3];
    outOrder[o] = (float)li;
}

// ---------------------------------------------------------------- greedy NMS + topk, writes final outputs
__global__ __launch_bounds__(1024)
void nms_kernel(const float* __restrict__ sboxes, const float* __restrict__ sscores,
                float* __restrict__ outBoxes, float* __restrict__ outScores,
                float* __restrict__ outKeep) {
    const int img = blockIdx.x;
    const int t   = threadIdx.x;
    __shared__ float X1s[MPROP], Y1s[MPROP], X2s[MPROP], Y2s[MPROP], AR[MPROP], SC[MPROP];
    __shared__ int   KP[MPROP];

    for (int j = t; j < MPROP; j += blockDim.x) {
        int g = img * MPROP + j;
        float a = sboxes[g * 4 + 0], b = sboxes[g * 4 + 1];
        float c = sboxes[g * 4 + 2], d = sboxes[g * 4 + 3];
        float s = sscores[g];
        X1s[j] = a; Y1s[j] = b; X2s[j] = c; Y2s[j] = d;
        AR[j]  = (c - a) * (d - b);
        SC[j]  = s;
        KP[j]  = (s > SCORE_THRESH) ? 1 : 0;
    }
    __syncthreads();

    for (int i = 0; i < MPROP - 1; ++i) {
        int ki = KP[i];
        if (ki) {
            float xi1 = X1s[i], yi1 = Y1s[i], xi2 = X2s[i], yi2 = Y2s[i], ai = AR[i];
            for (int j = i + 1 + t; j < MPROP; j += blockDim.x) {
                if (KP[j]) {
                    float xx1 = fmaxf(xi1, X1s[j]);
                    float yy1 = fmaxf(yi1, Y1s[j]);
                    float xx2 = fminf(xi2, X2s[j]);
                    float yy2 = fminf(yi2, Y2s[j]);
                    float iw = fmaxf(xx2 - xx1, 0.f), ih = fmaxf(yy2 - yy1, 0.f);
                    float inter = iw * ih;
                    float iou = inter / fmaxf(ai + AR[j] - inter, 1e-9f);
                    if (iou > NMS_THRESH) KP[j] = 0;
                }
            }
        }
        __syncthreads();
    }

    if (t == 0) {
        int run = 0;
        for (int j = 0; j < MPROP; ++j)
            if (KP[j]) { ++run; if (run > TOPK) KP[j] = 0; }
    }
    __syncthreads();

    for (int j = t; j < MPROP; j += blockDim.x) {
        int g = img * MPROP + j;
        outBoxes[g * 4 + 0] = X1s[j];
        outBoxes[g * 4 + 1] = Y1s[j];
        outBoxes[g * 4 + 2] = X2s[j];
        outBoxes[g * 4 + 3] = Y2s[j];
        outScores[g] = SC[j];
        outKeep[g]   = KP[j] ? 1.0f : 0.0f;
    }
}

// ---------------------------------------------------------------- launch
extern "C" void kernel_launch(void* const* d_in, const int* in_sizes, int n_in,
                              void* d_out, int out_size, void* d_ws, size_t ws_size,
                              hipStream_t stream) {
    const float* p2 = (const float*)d_in[0];
    const float* p3 = (const float*)d_in[1];
    const float* p4 = (const float*)d_in[2];
    const float* p5 = (const float*)d_in[3];
    const float* proposals = (const float*)d_in[4];
    const int*   image_sizes = (const int*)d_in[5];
    const float* W1 = (const float*)d_in[6];
    const float* b1 = (const float*)d_in[7];
    const float* W2 = (const float*)d_in[8];
    const float* b2 = (const float*)d_in[9];
    const float* Wc = (const float*)d_in[10];
    const float* bc = (const float*)d_in[11];
    const float* Wb = (const float*)d_in[12];
    const float* bb = (const float*)d_in[13];

    // workspace carve-up (256B aligned)
    char* base = (char*)d_ws;
    size_t off = 0;
    auto take = [&](size_t bytes) { size_t r = off; off += (bytes + 255) & ~(size_t)255; return r; };
    unsigned short* Xbf   = (unsigned short*)(base + take((size_t)NROWS * FCIN * 2));
    unsigned short* W1t   = (unsigned short*)(base + take((size_t)FCIN * FCDIM * 2));   // N x K
    unsigned short* W2t   = (unsigned short*)(base + take((size_t)FCDIM * FCDIM * 2));  // N x K
    unsigned short* X1bf  = (unsigned short*)(base + take((size_t)NROWS * FCDIM * 2));
    float*          X2f   = (float*)(base + take((size_t)NROWS * FCDIM * 4));
    float*          pbox  = (float*)(base + take((size_t)NROWS * 4 * 4));
    float*          pscr  = (float*)(base + take((size_t)NROWS * 4));
    float*          sbox  = (float*)(base + take((size_t)NROWS * 4 * 4));
    float*          sscr  = (float*)(base + take((size_t)NROWS * 4));
    (void)ws_size; (void)in_sizes; (void)n_in; (void)out_size;

    float* outBoxes  = (float*)d_out;                 // (2,1000,4)
    float* outScores = outBoxes + NROWS * 4;          // (2,1000)
    float* outKeep   = outScores + NROWS;             // (2,1000)
    float* outOrder  = outKeep + NROWS;               // (2,1000)

    // 1) weight conversion to bf16, transposed to N x K for contiguous B staging
    {
        int n1 = FCIN * FCDIM;
        cvt_transpose_bf16_kernel<<<(n1 + 255) / 256, 256, 0, stream>>>(W1, W1t, FCIN, FCDIM);
        int n2 = FCDIM * FCDIM;
        cvt_transpose_bf16_kernel<<<(n2 + 255) / 256, 256, 0, stream>>>(W2, W2t, FCDIM, FCDIM);
    }
    // 2) ROI pooling -> bf16 feature rows (2000 x 12544)
    pool_kernel<<<NROWS, 256, 0, stream>>>(p2, p3, p4, p5, proposals, Xbf);

    // 3) FC1: (2000 x 12544) * (12544 x 1024), relu -> bf16
    {
        dim3 g((NROWS + 127) / 128, FCDIM / 128);
        gemm_bf16_wmma<<<g, 256, 0, stream>>>(Xbf, W1t, b1, nullptr, X1bf,
                                              NROWS, FCDIM, FCIN);
    }
    // 4) FC2: (2000 x 1024) * (1024 x 1024), relu -> f32
    {
        dim3 g((NROWS + 127) / 128, FCDIM / 128);
        gemm_bf16_wmma<<<g, 256, 0, stream>>>(X1bf, W2t, b2, X2f, nullptr,
                                              NROWS, FCDIM, FCDIM);
    }
    // 5) heads + softmax + delta decode + clip
    heads_kernel<<<NROWS / 8, 256, 0, stream>>>(X2f, Wc, bc, Wb, bb, proposals,
                                                image_sizes, pbox, pscr);
    // 6) stable descending sort by rank
    sort_rank_kernel<<<(NROWS + 255) / 256, 256, 0, stream>>>(pbox, pscr, sbox, sscr, outOrder);
    // 7) greedy NMS + topk -> final outputs
    nms_kernel<<<NIMG, 1024, 0, stream>>>(sbox, sscr, outBoxes, outScores, outKeep);
}
